// AdaptiveLogSoftmaxWithLossAndIgnoreIndex_27530740367371
// MI455X (gfx1250) — compile-verified, matrix operand order checked by reading, not run on hardware
//
#include <hip/hip_runtime.h>
#include <hip/hip_bf16.h>

typedef __attribute__((ext_vector_type(16))) __bf16 v16bf;
typedef __attribute__((ext_vector_type(8)))  __bf16 v8bf;
typedef __attribute__((ext_vector_type(8)))  float  v8f;

#define M_ROWS  4096
#define K_IN    1024
#define HEAD_N  2002
#define HEAD_NP 2048   /* multiple of 64 */
#define H0      256
#define N0      8000
#define N0P     8000   /* 8000 = 125*64 */
#define H1      64
#define N1      40257
#define N1P     40320  /* 630*64 */
#define C0_LOW  2000
#define C1_LOW  10000
#define IGN     (-1)

__device__ __forceinline__ unsigned short f2bf(float f) {
  unsigned int u = __float_as_uint(f);
  u += 0x7fffu + ((u >> 16) & 1u);           // round-to-nearest-even
  return (unsigned short)(u >> 16);
}

// load one 16-bit WMMA operand fragment: 8 bf16 at p, 8 bf16 at p+16 (ISA 16x32 layout)
__device__ __forceinline__ v16bf ldfrag(const unsigned short* p) {
  v8bf lo = *(const v8bf*)p;
  v8bf hi = *(const v8bf*)(p + 16);
  return __builtin_shufflevector(lo, hi, 0,1,2,3,4,5,6,7,8,9,10,11,12,13,14,15);
}

// ---------- f32 -> bf16 conversion with zero tail-padding ----------
__global__ void cvt_bf16_pad(const float* __restrict__ src, unsigned short* __restrict__ dst,
                             long long nsrc, long long ndst) {
  long long i = (long long)blockIdx.x * blockDim.x + threadIdx.x;
  long long stride = (long long)gridDim.x * blockDim.x;
  for (; i < ndst; i += stride)
    dst[i] = (i < nsrc) ? f2bf(src[i]) : (unsigned short)0;
}

// ---------- WMMA GEMM: C[M,N] = A[M,K] * B[N,K]^T, outputs f32 + bf16 ----------
// small projection GEMMs (2.6 GFLOP total): one wave per 16x16 tile is fine
__global__ void wmma_gemm_out(const unsigned short* __restrict__ A,
                              const unsigned short* __restrict__ B,
                              float* __restrict__ Cf,
                              unsigned short* __restrict__ Cbf,
                              int K, int N) {
  const int rowBase = blockIdx.x * 16;
  const int colBase = blockIdx.y * 16;
  const int lane = threadIdx.x & 31;
  const int n16 = lane & 15;
  const int half = lane >> 4;
  const unsigned short* arow = A + (size_t)(rowBase + n16) * K;
  const unsigned short* brow = B + (size_t)(colBase + n16) * K;
  v8f c = {};
  for (int k0 = 0; k0 < K; k0 += 32) {
    const int kb = k0 + half * 8;
    v16bf a = ldfrag(arow + kb);
    v16bf b = ldfrag(brow + kb);
    c = __builtin_amdgcn_wmma_f32_16x16x32_bf16(false, a, false, b, (short)0, c, false, false);
  }
#pragma unroll
  for (int v = 0; v < 8; ++v) {
    const int row = rowBase + v + 8 * half;
    const int col = colBase + n16;
    const float val = c[v];
    Cf[(size_t)row * N + col] = val;
    Cbf[(size_t)row * N + col] = f2bf(val);
  }
}

// ---------- WMMA GEMM + fused streaming logsumexp over the N dimension ----------
// block = 32 rows (A panel in dynamic LDS), 8 waves stripe 64-column groups.
// Per wave: 2 row-tiles x 4 col-tiles register block -> 8 accumulators,
// A fragments reused 4x (from LDS), B fragments reused 2x (from global/L2).
__global__ void wmma_gemm_lse(const unsigned short* __restrict__ A,   // [M x K] bf16
                              const unsigned short* __restrict__ B,   // [Npad x K] bf16 zero-padded
                              int K, int Ncols, int Npad,
                              float* __restrict__ lse_out) {
  const int rowBase = blockIdx.x * 32;
  const int tid  = threadIdx.x;
  const int wave = tid >> 5;
  const int lane = tid & 31;
  const int nwaves = blockDim.x >> 5;
  const int n16  = lane & 15;
  const int half = lane >> 4;

  extern __shared__ __align__(16) char smem[];
  unsigned short* Alds = (unsigned short*)smem;          // 32 x K bf16
  __shared__ float Mred[8][32];
  __shared__ float Sred[8][32];

  { // stage the 32xK A panel into LDS (16B chunks)
    const uint4* src = (const uint4*)(A + (size_t)rowBase * K);
    uint4* dst = (uint4*)Alds;
    const int cnt = (32 * K) / 8;                        // 8 u16 per uint4
    for (int i = tid; i < cnt; i += blockDim.x) dst[i] = src[i];
  }
  __syncthreads();

  float m[2][8], s[2][8];
#pragma unroll
  for (int r = 0; r < 2; ++r)
#pragma unroll
    for (int v = 0; v < 8; ++v) { m[r][v] = -__builtin_inff(); s[r][v] = 0.f; }

  const int ngroups = Npad >> 6;                         // 64 columns per group
  const unsigned short* arow0 = Alds + (size_t)n16 * K;         // row tile 0
  const unsigned short* arow1 = Alds + (size_t)(16 + n16) * K;  // row tile 1

  for (int g = wave; g < ngroups; g += nwaves) {
    const int colBase = g << 6;
    const size_t brow0 = (size_t)(colBase + n16) * K;
    if (g + nwaves < ngroups)                            // prefetch next group's B
      __builtin_prefetch(B + brow0 + (((size_t)nwaves) << 6) * K, 0, 1);

    v8f c[2][4] = {};
    for (int k0 = 0; k0 < K; k0 += 32) {
      const int kb = k0 + half * 8;
      v16bf a0 = ldfrag(arow0 + kb);
      v16bf a1 = ldfrag(arow1 + kb);
#pragma unroll
      for (int j = 0; j < 4; ++j) {
        v16bf b = ldfrag(B + brow0 + (size_t)j * 16 * K + kb);
        c[0][j] = __builtin_amdgcn_wmma_f32_16x16x32_bf16(false, a0, false, b, (short)0, c[0][j], false, false);
        c[1][j] = __builtin_amdgcn_wmma_f32_16x16x32_bf16(false, a1, false, b, (short)0, c[1][j], false, false);
      }
    }
#pragma unroll
    for (int j = 0; j < 4; ++j) {
      if (colBase + j * 16 + n16 < Ncols) {              // mask zero-padded columns
#pragma unroll
        for (int r = 0; r < 2; ++r)
#pragma unroll
          for (int v = 0; v < 8; ++v) {                  // online logsumexp
            const float z = c[r][j][v];
            if (z > m[r][v]) { s[r][v] = s[r][v] * __expf(m[r][v] - z) + 1.0f; m[r][v] = z; }
            else             { s[r][v] += __expf(z - m[r][v]); }
          }
      }
    }
  }

  // merge across the 16 lanes of each half (C/D layout: lane = column)
#pragma unroll
  for (int mask = 1; mask <= 8; mask <<= 1) {
#pragma unroll
    for (int r = 0; r < 2; ++r)
#pragma unroll
      for (int v = 0; v < 8; ++v) {
        const float mo = __shfl_xor(m[r][v], mask, 32);
        const float so = __shfl_xor(s[r][v], mask, 32);
        const float Mx = fmaxf(m[r][v], mo);
        float sn = 0.f;
        if (Mx != -__builtin_inff())
          sn = s[r][v] * __expf(m[r][v] - Mx) + so * __expf(mo - Mx);
        m[r][v] = Mx; s[r][v] = sn;
      }
  }
  if (n16 == 0) {
#pragma unroll
    for (int r = 0; r < 2; ++r)
#pragma unroll
      for (int v = 0; v < 8; ++v) {
        const int ridx = r * 16 + half * 8 + v;
        Mred[wave][ridx] = m[r][v];
        Sred[wave][ridx] = s[r][v];
      }
  }
  __syncthreads();
  if (tid < 32) {                                        // merge across waves, finalize
    float Mv = -__builtin_inff(), Sv = 0.f;
    for (int w = 0; w < nwaves; ++w) {
      const float mo = Mred[w][tid], so = Sred[w][tid];
      const float Mx = fmaxf(Mv, mo);
      float sn = 0.f;
      if (Mx != -__builtin_inff()) sn = Sv * __expf(Mv - Mx) + so * __expf(mo - Mx);
      Mv = Mx; Sv = sn;
    }
    lse_out[rowBase + tid] = Mv + __logf(Sv);
  }
}

// ---------- picked head logits: z[t_head], z[2000], z[2001] per row ----------
__global__ void head_pick(const float* __restrict__ x, const float* __restrict__ hw,
                          const int* __restrict__ target,
                          float* __restrict__ zt, float* __restrict__ zc0, float* __restrict__ zc1) {
  const int row = blockIdx.x;
  const int t = target[row];
  int th = (t == IGN) ? 0 : t;
  if (th < 0) th = 0; if (th > C0_LOW - 1) th = C0_LOW - 1;
  const float* xr = x + (size_t)row * K_IN;
  const float* w0 = hw + (size_t)th * K_IN;
  const float* w1 = hw + (size_t)C0_LOW * K_IN;
  const float* w2 = hw + (size_t)(C0_LOW + 1) * K_IN;
  float d0 = 0.f, d1 = 0.f, d2 = 0.f;
  for (int k = threadIdx.x; k < K_IN; k += blockDim.x) {
    const float xv = xr[k];
    d0 += xv * w0[k]; d1 += xv * w1[k]; d2 += xv * w2[k];
  }
  for (int mask = 16; mask >= 1; mask >>= 1) {
    d0 += __shfl_xor(d0, mask, 32);
    d1 += __shfl_xor(d1, mask, 32);
    d2 += __shfl_xor(d2, mask, 32);
  }
  __shared__ float r0[4], r1[4], r2[4];
  const int wave = threadIdx.x >> 5, lane = threadIdx.x & 31;
  if (lane == 0) { r0[wave] = d0; r1[wave] = d1; r2[wave] = d2; }
  __syncthreads();
  if (threadIdx.x == 0) {
    float a = 0.f, b = 0.f, c = 0.f;
    for (int w = 0; w < (int)(blockDim.x >> 5); ++w) { a += r0[w]; b += r1[w]; c += r2[w]; }
    zt[row] = a; zc0[row] = b; zc1[row] = c;
  }
}

// ---------- picked cluster logit: dot(hidden[row], W[clip(t-low)]) ----------
__global__ void cluster_pick(const float* __restrict__ H, int Kh,
                             const float* __restrict__ W, int low, int high,
                             const int* __restrict__ target, float* __restrict__ z_out) {
  const int row = blockIdx.x;
  const int t = target[row];
  int rel = ((t == IGN) ? 0 : t) - low;
  if (rel < 0) rel = 0;
  const int mx = high - low - 1;
  if (rel > mx) rel = mx;
  const float* hr = H + (size_t)row * Kh;
  const float* wr = W + (size_t)rel * Kh;
  float d = 0.f;
  for (int k = threadIdx.x; k < Kh; k += blockDim.x) d += hr[k] * wr[k];
  for (int mask = 16; mask >= 1; mask >>= 1) d += __shfl_xor(d, mask, 32);
  __shared__ float r[2];
  const int wave = threadIdx.x >> 5, lane = threadIdx.x & 31;
  if (lane == 0) r[wave] = d;
  __syncthreads();
  if (threadIdx.x == 0) z_out[row] = r[0] + r[1];
}

// ---------- final masked-mean NLL ----------
__global__ void loss_kernel(const int* __restrict__ target,
                            const float* __restrict__ lseh, const float* __restrict__ zt,
                            const float* __restrict__ zc0,  const float* __restrict__ zc1,
                            const float* __restrict__ lse0, const float* __restrict__ z0,
                            const float* __restrict__ lse1, const float* __restrict__ z1,
                            float* __restrict__ out) {
  __shared__ float ssum[256];
  __shared__ float scnt[256];
  float sum = 0.f, cnt = 0.f;
  for (int r = threadIdx.x; r < M_ROWS; r += blockDim.x) {
    const int t = target[r];
    if (t == IGN) continue;
    const float lh = lseh[r];
    float lp;
    if (t < C0_LOW)      lp = zt[r] - lh;
    else if (t < C1_LOW) lp = (zc0[r] - lh) + (z0[r] - lse0[r]);
    else                 lp = (zc1[r] - lh) + (z1[r] - lse1[r]);
    sum += lp; cnt += 1.f;
  }
  ssum[threadIdx.x] = sum; scnt[threadIdx.x] = cnt;
  __syncthreads();
  for (int st = blockDim.x >> 1; st > 0; st >>= 1) {
    if ((int)threadIdx.x < st) {
      ssum[threadIdx.x] += ssum[threadIdx.x + st];
      scnt[threadIdx.x] += scnt[threadIdx.x + st];
    }
    __syncthreads();
  }
  if (threadIdx.x == 0) out[0] = -ssum[0] / scnt[0];
}

extern "C" void kernel_launch(void* const* d_in, const int* in_sizes, int n_in,
                              void* d_out, int out_size, void* d_ws, size_t ws_size,
                              hipStream_t stream) {
  (void)in_sizes; (void)n_in; (void)out_size; (void)ws_size;
  const float* x   = (const float*)d_in[0];
  const int*   tgt = (const int*)d_in[1];
  const float* hw  = (const float*)d_in[2];
  const float* p0  = (const float*)d_in[3];
  const float* o0  = (const float*)d_in[4];
  const float* p1  = (const float*)d_in[5];
  const float* o1  = (const float*)d_in[6];
  float* out = (float*)d_out;

  char* ws = (char*)d_ws;
  size_t off = 0;
  auto alloc = [&](size_t bytes) -> char* {
    char* p = ws + off;
    off = (off + bytes + 255) & ~(size_t)255;
    return p;
  };
  unsigned short* xbf  = (unsigned short*)alloc((size_t)M_ROWS * K_IN * 2);
  unsigned short* hwbf = (unsigned short*)alloc((size_t)HEAD_NP * K_IN * 2);
  unsigned short* p0bf = (unsigned short*)alloc((size_t)H0 * K_IN * 2);
  unsigned short* o0bf = (unsigned short*)alloc((size_t)N0P * H0 * 2);
  unsigned short* p1bf = (unsigned short*)alloc((size_t)H1 * K_IN * 2);
  unsigned short* o1bf = (unsigned short*)alloc((size_t)N1P * H1 * 2);
  float*          h0f  = (float*)alloc((size_t)M_ROWS * H0 * 4);
  unsigned short* h0bf = (unsigned short*)alloc((size_t)M_ROWS * H0 * 2);
  float*          h1f  = (float*)alloc((size_t)M_ROWS * H1 * 4);
  unsigned short* h1bf = (unsigned short*)alloc((size_t)M_ROWS * H1 * 2);
  float* lseh = (float*)alloc(M_ROWS * 4);
  float* lse0 = (float*)alloc(M_ROWS * 4);
  float* lse1 = (float*)alloc(M_ROWS * 4);
  float* zt   = (float*)alloc(M_ROWS * 4);
  float* zc0  = (float*)alloc(M_ROWS * 4);
  float* zc1  = (float*)alloc(M_ROWS * 4);
  float* z0   = (float*)alloc(M_ROWS * 4);
  float* z1   = (float*)alloc(M_ROWS * 4);

  // 1. bf16 conversion (+ zero pad for the N-padded weight matrices)
  cvt_bf16_pad<<<2048, 256, 0, stream>>>(x,  xbf,  (long long)M_ROWS * K_IN, (long long)M_ROWS * K_IN);
  cvt_bf16_pad<<<2048, 256, 0, stream>>>(hw, hwbf, (long long)HEAD_N * K_IN, (long long)HEAD_NP * K_IN);
  cvt_bf16_pad<<<512,  256, 0, stream>>>(p0, p0bf, (long long)H0 * K_IN,     (long long)H0 * K_IN);
  cvt_bf16_pad<<<2048, 256, 0, stream>>>(o0, o0bf, (long long)N0 * H0,       (long long)N0P * H0);
  cvt_bf16_pad<<<256,  256, 0, stream>>>(p1, p1bf, (long long)H1 * K_IN,     (long long)H1 * K_IN);
  cvt_bf16_pad<<<2048, 256, 0, stream>>>(o1, o1bf, (long long)N1 * H1,       (long long)N1P * H1);

  // 2. projections: hidden0 = x*proj0^T, hidden1 = x*proj1^T   (WMMA bf16)
  wmma_gemm_out<<<dim3(M_ROWS / 16, H0 / 16), 32, 0, stream>>>(xbf, p0bf, h0f, h0bf, K_IN, H0);
  wmma_gemm_out<<<dim3(M_ROWS / 16, H1 / 16), 32, 0, stream>>>(xbf, p1bf, h1f, h1bf, K_IN, H1);

  // 3. GEMM + fused streaming logsumexp (logit matrices never materialized)
  wmma_gemm_lse<<<M_ROWS / 32, 256, (size_t)32 * K_IN * 2, stream>>>(xbf,  hwbf, K_IN, HEAD_N, HEAD_NP, lseh);
  wmma_gemm_lse<<<M_ROWS / 32, 256, (size_t)32 * H0   * 2, stream>>>(h0bf, o0bf, H0,   N0,     N0P,     lse0);
  wmma_gemm_lse<<<M_ROWS / 32, 256, (size_t)32 * H1   * 2, stream>>>(h1bf, o1bf, H1,   N1,     N1P,     lse1);

  // 4. gather the per-row target logits (tiny dot products)
  head_pick<<<M_ROWS, 128, 0, stream>>>(x, hw, tgt, zt, zc0, zc1);
  cluster_pick<<<M_ROWS, 64, 0, stream>>>(h0f, H0, o0, C0_LOW, C1_LOW, tgt, z0);
  cluster_pick<<<M_ROWS, 64, 0, stream>>>(h1f, H1, o1, C1_LOW, 50257, tgt, z1);

  // 5. masked-mean NLL
  loss_kernel<<<1, 256, 0, stream>>>(tgt, lseh, zt, zc0, zc1, lse0, z0, lse1, z1, out);
}
